// Proposed_ver2_21071109554386
// MI455X (gfx1250) — matrix operand order repaired and checked by reading, not compiled, not run on hardware
//
#include <hip/hip_runtime.h>
#include <hip/hip_bf16.h>

// ---------------------------------------------------------------------------
// Problem constants (from reference): N=128, C=256, H=W=32, GROUP=32
// ---------------------------------------------------------------------------
#define NROWS   32768       // N*C
#define HW      1024        // H*W
#define NG      32          // groups
#define EPSV    1e-5f

typedef __attribute__((ext_vector_type(16))) __bf16 v16bf;
typedef __attribute__((ext_vector_type(8)))  float  v8f;

// Pack a float4 into 4 consecutive bf16 slots of a v16bf fragment.
__device__ __forceinline__ void pack4(v16bf& v, int base, float4 f) {
    v[base + 0] = (__bf16)f.x;
    v[base + 1] = (__bf16)f.y;
    v[base + 2] = (__bf16)f.z;
    v[base + 3] = (__bf16)f.w;
}

__device__ __forceinline__ void acc4(float& s, float& q, float4 f) {
    s += f.x + f.y + f.z + f.w;
    q = fmaf(f.x, f.x, q); q = fmaf(f.y, f.y, q);
    q = fmaf(f.z, f.z, q); q = fmaf(f.w, f.w, q);
}

// ---------------------------------------------------------------------------
// Kernel 1: per-16-row tile -> logits via WMMA bf16, row sum/sumsq,
//           argmax -> group id, atomic group accumulation.
// Block = 128 threads (4 waves). Wave w covers K in [w*256, w*256+256).
// ---------------------------------------------------------------------------
__global__ __launch_bounds__(128)
void k_stats(const float* __restrict__ x,
             const float* __restrict__ fc_w,
             const float* __restrict__ fc_b,
             int*   __restrict__ s_out,
             float* __restrict__ gacc)   // [3*NG]: cnt, sum, sumsq
{
    __shared__ float lds_logits[16 * NG];
    __shared__ float lds_rsum[16];
    __shared__ float lds_rsq[16];

    const int tid  = threadIdx.x;
    const int wave = tid >> 5;
    const int lane = tid & 31;
    const int m    = lane & 15;   // row within tile (A: M = lane%16)
    const int hi   = lane >> 4;   // half-wave selector

    for (int i = tid; i < 16 * NG; i += 128) lds_logits[i] = 0.0f;
    if (tid < 16) { lds_rsum[tid] = 0.0f; lds_rsq[tid] = 0.0f; }
    __syncthreads();

    const int rowbase = blockIdx.x * 16;
    const float* __restrict__ arow = x + (size_t)(rowbase + m) * HW;

    // A fragment (contiguous k order, not AMD's table order): lane holds K offsets
    //   {hi*16 .. hi*16+15} within each 32-wide chunk (matches B).
    const int ka = hi * 16;
    // B fragment (contiguous k order, same as A): lane holds N = lane%16,
    //   K = hi*16 + (0..15) consecutive -> 16 consecutive fc_w elements.
    const float* __restrict__ b0row = fc_w + (size_t)m        * HW + hi * 16;
    const float* __restrict__ b1row = fc_w + (size_t)(m + 16) * HW + hi * 16;

    v8f c0 = {}; v8f c1 = {};
    float rs = 0.0f, rq = 0.0f;

    const int k0 = wave * 256;
    #pragma unroll 2
    for (int k = k0; k < k0 + 256; k += 32) {
        // ---- A: 16 fp32 -> bf16 (and row statistics in fp32) ----
        float4 a0 = *(const float4*)(arow + k + ka);
        float4 a1 = *(const float4*)(arow + k + ka + 4);
        float4 a2 = *(const float4*)(arow + k + ka + 8);
        float4 a3 = *(const float4*)(arow + k + ka + 12);
        acc4(rs, rq, a0); acc4(rs, rq, a1); acc4(rs, rq, a2); acc4(rs, rq, a3);
        v16bf a;
        pack4(a, 0, a0); pack4(a, 4, a1); pack4(a, 8, a2); pack4(a, 12, a3);

        // ---- B: fc_w^T tiles for N-tile 0 (g 0..15) and 1 (g 16..31) ----
        v16bf b0, b1;
        pack4(b0, 0,  *(const float4*)(b0row + k));
        pack4(b0, 4,  *(const float4*)(b0row + k + 4));
        pack4(b0, 8,  *(const float4*)(b0row + k + 8));
        pack4(b0, 12, *(const float4*)(b0row + k + 12));
        pack4(b1, 0,  *(const float4*)(b1row + k));
        pack4(b1, 4,  *(const float4*)(b1row + k + 4));
        pack4(b1, 8,  *(const float4*)(b1row + k + 8));
        pack4(b1, 12, *(const float4*)(b1row + k + 12));

        // D = A x B + C   (v_wmma_f32_16x16x32_bf16)
        c0 = __builtin_amdgcn_wmma_f32_16x16x32_bf16(
                 false, a, false, b0, (short)0, c0, false, false);
        c1 = __builtin_amdgcn_wmma_f32_16x16x32_bf16(
                 false, a, false, b1, (short)0, c1, false, false);
    }

    // ---- merge per-lane stats (lane l and l+16 share row m; 4 waves) ----
    atomicAdd(&lds_rsum[m], rs);
    atomicAdd(&lds_rsq[m],  rq);

    // ---- merge logits partials across waves.
    // C layout: VGPR r, lane l -> M = r + 8*(l/16), N = l%16.
    #pragma unroll
    for (int r = 0; r < 8; ++r) {
        const int M = r + hi * 8;
        atomicAdd(&lds_logits[M * NG + m],      c0[r]);
        atomicAdd(&lds_logits[M * NG + 16 + m], c1[r]);
    }
    __syncthreads();

    // ---- argmax (first-max wins, matching jnp.argmax) + group atomics ----
    if (tid < 16) {
        const int row = tid;
        float best = -3.402823466e38f;
        int   bg   = 0;
        #pragma unroll
        for (int g = 0; g < NG; ++g) {
            const float v = lds_logits[row * NG + g] + fc_b[g];
            if (v > best) { best = v; bg = g; }
        }
        s_out[rowbase + row] = bg;
        atomicAdd(&gacc[bg],          (float)HW);
        atomicAdd(&gacc[NG + bg],     lds_rsum[row]);
        atomicAdd(&gacc[2 * NG + bg], lds_rsq[row]);
    }
}

// ---------------------------------------------------------------------------
// Kernel 2: group mean / rstd (unbiased var). cnt==0 folds to identity.
// ---------------------------------------------------------------------------
__global__ void k_finalize(const float* __restrict__ gacc,
                           float* __restrict__ gparams)  // [2*NG]: mean, rstd
{
    const int g = threadIdx.x;
    if (g >= NG) return;
    const float cnt = gacc[g];
    const float sum = gacc[NG + g];
    const float sq  = gacc[2 * NG + g];
    const float safe = fmaxf(cnt, 1.0f);
    float mean = sum / safe;
    const float var  = (sq - cnt * mean * mean) / fmaxf(cnt - 1.0f, 1.0f);
    float rstd = rsqrtf(var + EPSV);
    if (!(cnt > 0.0f)) { mean = 0.0f; rstd = 1.0f; }  // passthrough rows
    gparams[g]      = mean;
    gparams[NG + g] = rstd;
}

// ---------------------------------------------------------------------------
// Kernel 3: streamed normalize, float4 vectorized.
// out = (x - mean[s])*rstd[s]*weight[c] + bias[c]
// ---------------------------------------------------------------------------
__global__ __launch_bounds__(256)
void k_norm(const float4* __restrict__ x4,
            const float*  __restrict__ weight,
            const float*  __restrict__ bias,
            const int*    __restrict__ s,
            const float*  __restrict__ gparams,
            float4* __restrict__ out4)
{
    const int i = blockIdx.x * 256 + threadIdx.x;     // float4 index, < 2^23
    const int row = i >> 8;                           // HW/4 = 256 float4/row
    const int c   = row & 255;                        // C = 256
    const int g   = s[row];
    const float mean = gparams[g];
    const float rstd = gparams[NG + g];
    const float w    = weight[c];
    const float b    = bias[c];
    const float4 v = x4[i];
    float4 o;
    o.x = fmaf((v.x - mean) * rstd, w, b);
    o.y = fmaf((v.y - mean) * rstd, w, b);
    o.z = fmaf((v.z - mean) * rstd, w, b);
    o.w = fmaf((v.w - mean) * rstd, w, b);
    out4[i] = o;
}

// ---------------------------------------------------------------------------
// Launch. Inputs: x, weight, bias, fc_w, fc_b. Output: [N,C,H,W] fp32.
// Workspace: s[32768] int | gacc[96] float | gparams[64] float.
// ---------------------------------------------------------------------------
extern "C" void kernel_launch(void* const* d_in, const int* in_sizes, int n_in,
                              void* d_out, int out_size, void* d_ws, size_t ws_size,
                              hipStream_t stream)
{
    const float* x      = (const float*)d_in[0];
    const float* weight = (const float*)d_in[1];
    const float* bias   = (const float*)d_in[2];
    const float* fc_w   = (const float*)d_in[3];
    const float* fc_b   = (const float*)d_in[4];

    int*   s_ws     = (int*)d_ws;
    float* gacc     = (float*)((char*)d_ws + (size_t)NROWS * sizeof(int));
    float* gparams  = gacc + 3 * NG;

    // zero the 3*NG group accumulators (capture-safe async memset)
    hipMemsetAsync(gacc, 0, 3 * NG * sizeof(float), stream);

    k_stats<<<NROWS / 16, 128, 0, stream>>>(x, fc_w, fc_b, s_ws, gacc);
    k_finalize<<<1, NG, 0, stream>>>(gacc, gparams);

    const int nvec = NROWS * (HW / 4);                // 8,388,608 float4
    k_norm<<<nvec / 256, 256, 0, stream>>>((const float4*)x, weight, bias,
                                           s_ws, gparams, (float4*)d_out);
}
